// TransformerBlockQuantum_65481071406507
// MI455X (gfx1250) — matrix operand order repaired and checked by reading, not compiled, not run on hardware
//
#include <hip/hip_runtime.h>

// ---------------------------------------------------------------------------
// TransformerBlockQuantum for MI455X (gfx1250, wave32, WMMA + TDM)
//
// _quantum_project reduces analytically to a prefix product of cosines within
// each 8-element chunk (CNOT chain = basis permutation b_k -> b_0^...^b_k;
// product-state marginals factorize). Heavy work = 4 GEMMs on
// v_wmma_f32_16x16x32_f16, with A/B tiles DMA'd into LDS by the Tensor Data
// Mover via inline asm (portable across toolchains; bypasses the
// arity-divergent clang builtin). Double-buffered LDS, wave 0 drives the DMA,
// sync via s_wait_tensorcnt + workgroup barrier.
// ---------------------------------------------------------------------------

typedef __attribute__((ext_vector_type(16))) _Float16 v16h;
typedef __attribute__((ext_vector_type(8)))  float    v8f;
typedef __attribute__((ext_vector_type(4)))  unsigned int v4u;
typedef __attribute__((ext_vector_type(8)))  int      v8i;

struct alignas(16) U4 { unsigned int a, b, c, d; };
struct alignas(32) Frag2 { U4 lo, hi; };
struct alignas(16) F4 { float x, y, z, w; };

__device__ __forceinline__ _Float16 f2h(float v) { return (_Float16)v; }

#define USE_TDM 1

#if __has_builtin(__builtin_amdgcn_s_wait_tensorcnt)
#define WAIT_TENSORCNT(n) __builtin_amdgcn_s_wait_tensorcnt(n)
#else
#define WAIT_TENSORCNT(n) asm volatile("s_wait_tensorcnt %0" ::"i"(n) : "memory")
#endif

// ---------------- problem sizes ----------------
constexpr int BB = 4, SS = 512, EE = 512, HH = 64, FF = 2048;
constexpr int MM = BB * SS;                       // 2048 tokens
constexpr float SCALE = 0.35355339059327373f;     // 1/sqrt(8)

// ---------------- workspace layout (bytes) ----------------
constexpr size_t OFF_PROJ = 0;
constexpr size_t SZ_PROJ  = (size_t)MM * EE * 2;
constexpr size_t OFF_WQKV = OFF_PROJ + SZ_PROJ;
constexpr size_t SZ_WQKV  = (size_t)3 * EE * EE * 2;
constexpr size_t OFF_WOUT = OFF_WQKV + SZ_WQKV;
constexpr size_t SZ_WOUT  = (size_t)EE * EE * 2;
constexpr size_t OFF_W1   = OFF_WOUT + SZ_WOUT;
constexpr size_t SZ_W1    = (size_t)FF * EE * 2;
constexpr size_t OFF_W2   = OFF_W1 + SZ_W1;
constexpr size_t SZ_W2    = (size_t)EE * FF * 2;
constexpr size_t OFF_QKV  = OFF_W2 + SZ_W2;
constexpr size_t SZ_QKV   = (size_t)MM * 3 * EE * 4;
constexpr size_t OFF_CTX  = OFF_QKV + SZ_QKV;
constexpr size_t SZ_CTX   = (size_t)MM * EE * 2;
constexpr size_t OFF_AOUT = OFF_CTX + SZ_CTX;
constexpr size_t SZ_AOUT  = (size_t)MM * EE * 4;
constexpr size_t OFF_X1F  = OFF_AOUT + SZ_AOUT;
constexpr size_t SZ_X1F   = SZ_AOUT;
constexpr size_t OFF_X1H  = OFF_X1F + SZ_X1F;
constexpr size_t SZ_X1H   = (size_t)MM * EE * 2;
constexpr size_t OFF_HID  = OFF_X1H + SZ_X1H;
constexpr size_t SZ_HID   = (size_t)MM * FF * 2;
constexpr size_t OFF_FFN  = OFF_HID + SZ_HID;

// ---------------------------------------------------------------------------
__global__ void cvt_f16_kernel(const float* __restrict__ src,
                               _Float16* __restrict__ dst, int n) {
  int i = blockIdx.x * 256 + threadIdx.x;
  if (i < n) dst[i] = f2h(src[i]);
}

// ---------------------------------------------------------------------------
// Quantum projection: cumulative product of cos within groups of 8
// ---------------------------------------------------------------------------
__global__ void quantum_proj_kernel(const float* __restrict__ x,
                                    _Float16* __restrict__ proj, int ngroups) {
  int g = blockIdx.x * 256 + threadIdx.x;
  if (g >= ngroups) return;
  const float* p = x + (size_t)g * 8;
  F4 a = *(const F4*)p;
  F4 b = *(const F4*)(p + 4);
  float v[8] = {a.x, a.y, a.z, a.w, b.x, b.y, b.z, b.w};
  alignas(16) _Float16 o[8];
  float prod = 1.0f;
#pragma unroll
  for (int j = 0; j < 8; ++j) {
    prod *= __cosf(v[j]);
    o[j] = f2h(prod);
  }
  *(U4*)&proj[(size_t)g * 8] = *(const U4*)o;
}

// ---------------------------------------------------------------------------
// TDM: DMA one 128x32 f16 tile (global row stride = rowStrideElems) into LDS
// at ldsOff, inserting 32B of padding after every 64B row -> LDS row stride
// 96B (= 48 f16), matching the compute-side layout. D# per ISA ch.8; 2D
// tensor -> 2-group form (VADDR2/3 = NULL). All operands wave-uniform.
// ---------------------------------------------------------------------------
#if USE_TDM
__device__ __forceinline__ void tdm_tile_load(const _Float16* g, unsigned ldsOff,
                                              int rowStrideElems) {
  unsigned long long ga = (unsigned long long)(uintptr_t)g;
  v4u g0;
  g0[0] = 1u;                                    // count=1, user descriptor
  g0[1] = ldsOff;                                // lds_addr (bytes)
  g0[2] = (unsigned)(ga & 0xFFFFFFFFu);          // global_addr[31:0]
  g0[3] = (unsigned)((ga >> 32) & 0x01FFFFFFu)   // global_addr[56:32]
          | 0x80000000u;                         // type=2 ("image")
  v8i g1;
  g1[0] = (int)((1u << 16)                       // data_size=1 -> 2 bytes
              | (1u << 20)                       // pad_enable
              | (3u << 22)                       // pad_interval: 64B
              | (7u << 25));                     // pad_amount: 8 dwords = 32B
  g1[1] = (int)(((unsigned)rowStrideElems & 0xFFFFu) << 16); // tensor_dim0 lo16
  g1[2] = (int)((((unsigned)rowStrideElems >> 16) & 0xFFFFu) // tensor_dim0 hi16
              | 0xFFFF0000u);                    // tensor_dim1 lo16 = 0xFFFF
  g1[3] = (int)(32u << 16);                      // tensor_dim1 hi=0, tile_dim0=32
  g1[4] = 128;                                   // tile_dim1=128, tile_dim2=0
  g1[5] = rowStrideElems;                        // tensor_dim0_stride[31:0]
  g1[6] = 0;                                     // stride0 hi, stride1 lo
  g1[7] = 0;
  asm volatile("tensor_load_to_lds %0, %1" ::"s"(g0), "s"(g1) : "memory");
}
#endif

// ---------------------------------------------------------------------------
// Tiled WMMA GEMM:  C[M,N] = A[M,K] * B[N,K]^T + bias[N]   (A,B f16 row-major)
// Block tile 128x128, K-step 32. 256 threads = 8 waves, each wave 32x64
// (2x4 WMMA fragments).
// ---------------------------------------------------------------------------
#define LDSS 48   // padded LDS row stride in f16 elements (96B)

template <bool OUTF32, bool RELU>
__global__ __launch_bounds__(256)
void gemm_f16_wmma_kernel(const _Float16* __restrict__ A,
                          const _Float16* __restrict__ B,
                          const float* __restrict__ bias,
                          float* __restrict__ outF,
                          _Float16* __restrict__ outH,
                          int M, int N, int K) {
#if USE_TDM
  __shared__ _Float16 lA[2][128 * LDSS];
  __shared__ _Float16 lB[2][128 * LDSS];
#else
  __shared__ _Float16 lA[1][128 * LDSS];
  __shared__ _Float16 lB[1][128 * LDSS];
#endif

  const int tid  = threadIdx.x;
  const int m0   = blockIdx.y * 128;
  const int n0   = blockIdx.x * 128;
  const int lane = tid & 31;
  const int wave = tid >> 5;
  const int wm   = wave & 3;
  const int wn   = wave >> 2;
  const int mr   = lane & 15;
  const int hi   = lane >> 4;

  v8f acc[2][4];
#pragma unroll
  for (int i = 0; i < 2; ++i)
#pragma unroll
    for (int j = 0; j < 4; ++j)
      acc[i][j] = (v8f){0.f, 0.f, 0.f, 0.f, 0.f, 0.f, 0.f, 0.f};

  const int nk = K >> 5;

#if USE_TDM
  // Prologue: wave 0 launches the DMA pair for tile 0.
  if (wave == 0) {
    tdm_tile_load(&A[(size_t)m0 * K], (unsigned)(uintptr_t)&lA[0][0], K);
    tdm_tile_load(&B[(size_t)n0 * K], (unsigned)(uintptr_t)&lB[0][0], K);
  }
  for (int ks = 0; ks < nk; ++ks) {
    const int cur = ks & 1;
    if (wave == 0) {
      if (ks + 1 < nk) {
        const int nxt = (ks + 1) & 1;
        const int k1 = (ks + 1) << 5;
        tdm_tile_load(&A[(size_t)m0 * K + k1], (unsigned)(uintptr_t)&lA[nxt][0], K);
        tdm_tile_load(&B[(size_t)n0 * K + k1], (unsigned)(uintptr_t)&lB[nxt][0], K);
        WAIT_TENSORCNT(2);   // in-order: current pair complete
      } else {
        WAIT_TENSORCNT(0);
      }
    }
    __syncthreads();   // all waves: current tile visible in LDS
    const _Float16* bufA = &lA[cur][0];
    const _Float16* bufB = &lB[cur][0];
#else
  for (int ks = 0; ks < nk; ++ks) {
    const int k0 = ks << 5;
    __syncthreads();
#pragma unroll
    for (int it = 0; it < 2; ++it) {
      int id  = tid + it * 256;
      int row = id >> 2;
      int c   = (id & 3) << 3;
      *(U4*)&lA[0][row * LDSS + c] = *(const U4*)&A[(size_t)(m0 + row) * K + k0 + c];
      *(U4*)&lB[0][row * LDSS + c] = *(const U4*)&B[(size_t)(n0 + row) * K + k0 + c];
    }
    __syncthreads();
    const _Float16* bufA = &lA[0][0];
    const _Float16* bufB = &lB[0][0];
#endif

    // A fragment (16x32 f16): lanes 0-15 hold K 0-7 & 16-23, lanes 16-31
    // hold K 8-15 & 24-31 (ISA 7.12.2).
    v16h afrag[2];
#pragma unroll
    for (int i = 0; i < 2; ++i) {
      int r = wm * 32 + i * 16 + mr;
      Frag2 f;
      f.lo = *(const U4*)&bufA[r * LDSS + hi * 8];
      f.hi = *(const U4*)&bufA[r * LDSS + 16 + hi * 8];
      afrag[i] = __builtin_bit_cast(v16h, f);
    }
    // B fragment (32x16 f16): lanes 0-15 hold K 0-7 & 16-23, lanes 16-31
    // hold K 8-15 & 24-31 (same as A).
    v16h bfrag[4];
#pragma unroll
    for (int j = 0; j < 4; ++j) {
      int r = wn * 64 + j * 16 + mr;
      Frag2 f;
      f.lo = *(const U4*)&bufB[r * LDSS + hi * 8];
      f.hi = *(const U4*)&bufB[r * LDSS + 16 + hi * 8];
      bfrag[j] = __builtin_bit_cast(v16h, f);
    }

#pragma unroll
    for (int i = 0; i < 2; ++i)
#pragma unroll
      for (int j = 0; j < 4; ++j)
        acc[i][j] = __builtin_amdgcn_wmma_f32_16x16x32_f16(
            false, afrag[i], false, bfrag[j], (short)0, acc[i][j], false, false);

    __syncthreads();   // tile consumed; safe to overwrite
  }

  // Branchless epilogue (template-selected output path).
#pragma unroll
  for (int i = 0; i < 2; ++i) {
#pragma unroll
    for (int j = 0; j < 4; ++j) {
      const int col = n0 + wn * 64 + j * 16 + mr;
      const int rb  = m0 + wm * 32 + i * 16 + hi * 8;
      const float bv = bias[col];
#pragma unroll
      for (int r = 0; r < 8; ++r) {
        float v = acc[i][j][r] + bv;
        if (RELU) v = fmaxf(v, 0.0f);
        const size_t idx = (size_t)(rb + r) * N + col;
        if (OUTF32) outF[idx] = v;
        else        outH[idx] = f2h(v);
      }
    }
  }
}

// ---------------------------------------------------------------------------
// Attention: per (b, h, 256-row q-slab); head dim 8 in registers, online
// softmax over S=512 keys streamed through LDS tiles of 64 rows.
// ---------------------------------------------------------------------------
__global__ __launch_bounds__(256)
void attention_kernel(const float* __restrict__ qkv,
                      _Float16* __restrict__ ctx) {
  __shared__ float Kt[64][8];
  __shared__ float Vt[64][8];

  const int blk   = blockIdx.x;
  const int qhalf = blk & 1;
  const int bh    = blk >> 1;
  const int b     = bh >> 6;
  const int h     = bh & 63;
  const int tid   = threadIdx.x;
  const int t     = b * SS + qhalf * 256 + tid;

  const float* qp = qkv + (size_t)t * (3 * EE) + h * 8;
  float q[8];
#pragma unroll
  for (int d = 0; d < 8; ++d) q[d] = qp[d] * SCALE;

  float mval = -1e30f, lsum = 0.0f;
  float accd[8] = {0.f, 0.f, 0.f, 0.f, 0.f, 0.f, 0.f, 0.f};

  const int lt  = tid & 127;
  const int lr  = lt >> 1;
  const int lh  = (lt & 1) * 4;
  const int sec = (tid < 128) ? EE : 2 * EE;

  for (int tile = 0; tile < SS / 64; ++tile) {
    int kb = b * SS + tile * 64;
    __syncthreads();
    F4 val = *(const F4*)(qkv + (size_t)(kb + lr) * (3 * EE) + sec + h * 8 + lh);
    if (tid < 128) *(F4*)&Kt[lr][lh] = val;
    else           *(F4*)&Vt[lr][lh] = val;
    __syncthreads();

#pragma unroll 4
    for (int kk = 0; kk < 64; ++kk) {
      float s = q[0] * Kt[kk][0];
#pragma unroll
      for (int d = 1; d < 8; ++d) s += q[d] * Kt[kk][d];
      float mn   = fmaxf(mval, s);
      float corr = __expf(mval - mn);
      float p    = __expf(s - mn);
      lsum = lsum * corr + p;
#pragma unroll
      for (int d = 0; d < 8; ++d) accd[d] = accd[d] * corr + p * Vt[kk][d];
      mval = mn;
    }
  }

  float inv = 1.0f / lsum;
  alignas(16) _Float16 o[8];
#pragma unroll
  for (int d = 0; d < 8; ++d) o[d] = f2h(accd[d] * inv);
  *(U4*)&ctx[(size_t)t * EE + h * 8] = *(const U4*)o;
}

// ---------------------------------------------------------------------------
// Fused residual add + LayerNorm. One block per token, 128 threads x 4 elems.
// ---------------------------------------------------------------------------
template <bool WRITE_H>
__global__ __launch_bounds__(128)
void add_ln_kernel(const float* __restrict__ xa,
                   const float* __restrict__ xb,
                   const float* __restrict__ w,
                   const float* __restrict__ bias,
                   float* __restrict__ outF,
                   _Float16* __restrict__ outH) {
  __shared__ float r1[128];
  __shared__ float r2[128];
  const int t = blockIdx.x;
  const int tid = threadIdx.x;
  const int c = tid * 4;

  F4 va = *(const F4*)(xa + (size_t)t * EE + c);
  F4 vb = *(const F4*)(xb + (size_t)t * EE + c);
  float v0 = va.x + vb.x, v1 = va.y + vb.y, v2 = va.z + vb.z, v3 = va.w + vb.w;

  r1[tid] = v0 + v1 + v2 + v3;
  r2[tid] = v0 * v0 + v1 * v1 + v2 * v2 + v3 * v3;
  __syncthreads();
  for (int off = 64; off > 0; off >>= 1) {
    if (tid < off) { r1[tid] += r1[tid + off]; r2[tid] += r2[tid + off]; }
    __syncthreads();
  }
  float mean = r1[0] * (1.0f / EE);
  float var  = r2[0] * (1.0f / EE) - mean * mean;
  float rstd = rsqrtf(var + 1e-5f);

  F4 wv = *(const F4*)(w + c);
  F4 bv = *(const F4*)(bias + c);
  F4 y;
  y.x = (v0 - mean) * rstd * wv.x + bv.x;
  y.y = (v1 - mean) * rstd * wv.y + bv.y;
  y.z = (v2 - mean) * rstd * wv.z + bv.z;
  y.w = (v3 - mean) * rstd * wv.w + bv.w;
  *(F4*)(outF + (size_t)t * EE + c) = y;
  if (WRITE_H) {
    _Float16* oh = outH + (size_t)t * EE + c;
    oh[0] = f2h(y.x); oh[1] = f2h(y.y); oh[2] = f2h(y.z); oh[3] = f2h(y.w);
  }
}

// ---------------------------------------------------------------------------
extern "C" void kernel_launch(void* const* d_in, const int* in_sizes, int n_in,
                              void* d_out, int out_size, void* d_ws, size_t ws_size,
                              hipStream_t stream) {
  const float* x         = (const float*)d_in[0];
  const float* in_proj_w = (const float*)d_in[1];
  const float* in_proj_b = (const float*)d_in[2];
  const float* out_w     = (const float*)d_in[3];
  const float* out_b     = (const float*)d_in[4];
  const float* ln1_w     = (const float*)d_in[5];
  const float* ln1_b     = (const float*)d_in[6];
  const float* ln2_w     = (const float*)d_in[7];
  const float* ln2_b     = (const float*)d_in[8];
  const float* w1        = (const float*)d_in[9];
  const float* b1        = (const float*)d_in[10];
  const float* w2        = (const float*)d_in[11];
  const float* b2        = (const float*)d_in[12];

  char* ws = (char*)d_ws;
  _Float16* projH = (_Float16*)(ws + OFF_PROJ);
  _Float16* wqkvH = (_Float16*)(ws + OFF_WQKV);
  _Float16* woutH = (_Float16*)(ws + OFF_WOUT);
  _Float16* w1H   = (_Float16*)(ws + OFF_W1);
  _Float16* w2H   = (_Float16*)(ws + OFF_W2);
  float*    qkvF  = (float*)(ws + OFF_QKV);
  _Float16* ctxH  = (_Float16*)(ws + OFF_CTX);
  float*    aoutF = (float*)(ws + OFF_AOUT);
  float*    x1F   = (float*)(ws + OFF_X1F);
  _Float16* x1H   = (_Float16*)(ws + OFF_X1H);
  _Float16* hidH  = (_Float16*)(ws + OFF_HID);
  float*    ffnF  = (float*)(ws + OFF_FFN);

  cvt_f16_kernel<<<(3 * EE * EE) / 256, 256, 0, stream>>>(in_proj_w, wqkvH, 3 * EE * EE);
  cvt_f16_kernel<<<(EE * EE) / 256, 256, 0, stream>>>(out_w, woutH, EE * EE);
  cvt_f16_kernel<<<(FF * EE) / 256, 256, 0, stream>>>(w1, w1H, FF * EE);
  cvt_f16_kernel<<<(EE * FF) / 256, 256, 0, stream>>>(w2, w2H, EE * FF);

  const int ngroups = MM * EE / 8;
  quantum_proj_kernel<<<ngroups / 256, 256, 0, stream>>>(x, projH, ngroups);

  // QKV = proj @ Wqkv^T + b     (M=2048, N=1536, K=512) -> f32
  gemm_f16_wmma_kernel<true, false><<<dim3(3 * EE / 128, MM / 128), 256, 0, stream>>>(
      projH, wqkvH, in_proj_b, qkvF, nullptr, MM, 3 * EE, EE);

  attention_kernel<<<BB * HH * 2, 256, 0, stream>>>(qkvF, ctxH);

  // attn_out = ctx @ Wout^T + b (M=2048, N=512, K=512) -> f32
  gemm_f16_wmma_kernel<true, false><<<dim3(EE / 128, MM / 128), 256, 0, stream>>>(
      ctxH, woutH, out_b, aoutF, nullptr, MM, EE, EE);

  // x1 = LN(x + attn_out)
  add_ln_kernel<true><<<MM, 128, 0, stream>>>(x, aoutF, ln1_w, ln1_b, x1F, x1H);

  // hid = relu(x1 @ W1^T + b1)  (M=2048, N=2048, K=512) -> f16
  gemm_f16_wmma_kernel<false, true><<<dim3(FF / 128, MM / 128), 256, 0, stream>>>(
      x1H, w1H, b1, nullptr, hidH, MM, FF, EE);

  // ffn = hid @ W2^T + b2       (M=2048, N=512, K=2048) -> f32
  gemm_f16_wmma_kernel<true, false><<<dim3(EE / 128, MM / 128), 256, 0, stream>>>(
      hidH, w2H, b2, ffnF, nullptr, MM, EE, FF);

  // out = LN(x1 + ffn)
  add_ln_kernel<false><<<MM, 128, 0, stream>>>(x1F, ffnF, ln2_w, ln2_b,
                                               (float*)d_out, nullptr);
}
